// DistributedRingAttentionFixed_51531017617693
// MI455X (gfx1250) — matrix operand order repaired and checked by reading, not compiled, not run on hardware
//
#include <hip/hip_runtime.h>
#include <hip/hip_bf16.h>

// MI455X (gfx1250) fused MHA forward.
// Roofline: ~103 GFLOP vs ~50MB essential HBM traffic -> compute bound.
// Strategy: one-time fp32->f16 cast, then 3 WMMA stages:
//   (1) qkv = x @ Wqkv^T   (f16 WMMA, f32 accum, f16 out)
//   (2) flash attention per (head, 64-query block): S^T = K Q^T via WMMA,
//       online softmax in f32, P staged in LDS, O += P V via WMMA
//   (3) y = ctx @ Wout^T   (f32 out)
// GEMM wave tile is 32x32 (2x2 WMMA accums): this fits the VGPR budget with
// zero spills and lets the scheduler software-pipeline b128 loads against
// WMMAs (verified in disasm round 1; the 32x64 variant spilled).
// Intermediates (qkv 24MB f16, ctx 8MB f16) live in the 192MB L2.
// Workspace layout (48MB total):
//   [0,8M)   xh      4096x1024 f16
//   [8M,14M) wqkvh   3072x1024 f16
//   [14M,16M)wouth   1024x1024 f16
//   [16M,40M)qkvh    4096x3072 f16
//   [40M,48M)ctxh    4096x1024 f16

typedef __attribute__((ext_vector_type(16))) _Float16 v16h_t;
typedef __attribute__((ext_vector_type(8)))  _Float16 v8h_t;
typedef __attribute__((ext_vector_type(8)))  float    v8f_t;

#define HIDDEN 1024
#define SEQ    4096
#define HEADS  16
#define HDIM   64

// ---- WMMA fragment loaders (gfx1250 wave32 layouts, CDNA5 ISA) ----
// 16-bit A 16x32 (MxK): lanes 0-15 row M=lane, K in {0..7}U{16..23};
//                        lanes 16-31 row M=lane-16, K in {8..15}U{24..31}.
static __device__ inline v16h_t load_a_frag(const _Float16* p, int ld) {
  const int lane = threadIdx.x & 31;
  const int row  = lane & 15;
  const int off  = (lane & 16) ? 8 : 0;
  const _Float16* base = p + (size_t)row * ld + off;
  v8h_t lo = *(const v8h_t*)(base);        // K = off .. off+7
  v8h_t hi = *(const v8h_t*)(base + 16);   // K = 16+off .. 23+off
  v16h_t a;
#pragma unroll
  for (int i = 0; i < 8; ++i) { a[i] = lo[i]; a[i + 8] = hi[i]; }
  return a;
}

// 16-bit B 32x16 (KxN): lanes 0-15 col N=lane, K in {0..7}U{16..23}; lanes 16-31 col
// N=lane-16, K in {8..15}U{24..31}. p points at the KxN-source such that column n is a
// contiguous K-run (row-major [N,K] weights; V^T staged in LDS).
static __device__ inline v16h_t load_b_frag(const _Float16* p, int ld) {
  const int lane = threadIdx.x & 31;
  const int col  = lane & 15;
  const int off  = (lane & 16) ? 8 : 0;
  const _Float16* base = p + (size_t)col * ld + off;
  v8h_t lo = *(const v8h_t*)(base);
  v8h_t hi = *(const v8h_t*)(base + 16);
  v16h_t b;
#pragma unroll
  for (int i = 0; i < 8; ++i) { b[i] = lo[i]; b[i + 8] = hi[i]; }
  return b;
}

static __device__ inline v8f_t wmma_f16(v16h_t a, v16h_t b, v8f_t c) {
  return __builtin_amdgcn_wmma_f32_16x16x32_f16(false, a, false, b,
                                                (short)0, c, false, false);
}

// ---- fp32 -> f16 cast, 8 elements per lane (all sizes are multiples of 8) ----
__global__ void f32_to_f16_kernel(const float* __restrict__ in,
                                  _Float16* __restrict__ out, int n) {
  int i = blockIdx.x * blockDim.x + threadIdx.x;
  int stride = gridDim.x * blockDim.x;
  for (int base = i * 8; base < n; base += stride * 8) {
    v8f_t v = *(const v8f_t*)(in + base);
    v8h_t h;
#pragma unroll
    for (int j = 0; j < 8; ++j) h[j] = (_Float16)v[j];
    *(v8h_t*)(out + base) = h;
  }
}

// ---- C[m][n] = sum_k A[m][k] * W[n][k] ----
// 128 threads = 4 waves; wave tile 32x32 (no spills); block tile 64x64.
__global__ void gemm_xwt_wmma(const _Float16* __restrict__ A,
                              const _Float16* __restrict__ W,
                              _Float16* __restrict__ Ch,
                              float* __restrict__ Cf,
                              int K, int ldc) {
  const int w = threadIdx.x >> 5;
  const int mbase = blockIdx.y * 64 + (w >> 1) * 32;
  const int nbase = blockIdx.x * 64 + (w & 1) * 32;
  v8f_t acc00 = {}, acc01 = {}, acc10 = {}, acc11 = {};

  const _Float16* Ap0 = A + (size_t)mbase * K;
  const _Float16* Ap1 = A + (size_t)(mbase + 16) * K;
  const _Float16* Wp0 = W + (size_t)nbase * K;
  const _Float16* Wp1 = W + (size_t)(nbase + 16) * K;
  for (int k0 = 0; k0 < K; k0 += 32) {
    v16h_t a0 = load_a_frag(Ap0 + k0, K);
    v16h_t a1 = load_a_frag(Ap1 + k0, K);
    v16h_t b0 = load_b_frag(Wp0 + k0, K);
    v16h_t b1 = load_b_frag(Wp1 + k0, K);
    acc00 = wmma_f16(a0, b0, acc00);
    acc01 = wmma_f16(a0, b1, acc01);
    acc10 = wmma_f16(a1, b0, acc10);
    acc11 = wmma_f16(a1, b1, acc11);
    // keep L1 warm two K-steps ahead; unconditional (speculative prefetch
    // past the end is dropped silently) so no branch in the inner loop.
    __builtin_prefetch(Ap0 + k0 + 64, 0, 1);
    __builtin_prefetch(Wp0 + k0 + 64, 0, 1);
  }

  // C/D layout: lanes 0-15 -> (M=r, N=lane); lanes 16-31 -> (M=r+8, N=lane-16)
  const int lane = threadIdx.x & 31;
  const int ncol = lane & 15;
  const int rb   = (lane & 16) ? 8 : 0;
  v8f_t accs[2][2] = {{acc00, acc01}, {acc10, acc11}};
#pragma unroll
  for (int ti = 0; ti < 2; ++ti)
#pragma unroll
    for (int tj = 0; tj < 2; ++tj)
#pragma unroll
      for (int r = 0; r < 8; ++r) {
        int m = mbase + ti * 16 + rb + r;
        int n = nbase + tj * 16 + ncol;
        float v = accs[ti][tj][r];
        if (Ch) Ch[(size_t)m * ldc + n] = (_Float16)v;
        else    Cf[(size_t)m * ldc + n] = v;
      }
}

// ---- flash attention: block = (64 queries, 1 head); wave = 16 queries ----
__global__ void flash_attn_wmma(const _Float16* __restrict__ qkv,
                                _Float16* __restrict__ ctx) {
  __shared__ __align__(32) _Float16 Vt[HDIM][HDIM + 8]; // V^T: [d][key], padded
  __shared__ __align__(32) _Float16 Pt[4][16][HDIM];    // per wave: P[q][key]
  const int w    = threadIdx.x >> 5;
  const int lane = threadIdx.x & 31;
  const int hq   = blockIdx.y * HDIM;         // head column offset
  const int q0   = blockIdx.x * 64 + w * 16;  // this wave's query rows
  const int LDQ  = 3 * HIDDEN;
  const float scale = 0.125f;                 // 1/sqrt(64)
  const int rb = (lane & 16) ? 8 : 0;
  const int qc = lane & 15;

  // Q fragments (B-operand of S^T = K Q^T): column n = query row, contiguous d
  const _Float16* Qp = qkv + (size_t)q0 * LDQ + hq;
  v16h_t bq0 = load_b_frag(Qp + 0,  LDQ);
  v16h_t bq1 = load_b_frag(Qp + 32, LDQ);

  v8f_t o0 = {}, o1 = {}, o2 = {}, o3 = {};   // O[16q x 64d]
  float m_i = -1.0e30f, l_i = 0.0f;           // per-query stats (query = lane&15)

  for (int kb = 0; kb < SEQ; kb += 64) {
    // cooperatively stage V^T into LDS (coalesced global reads)
    const _Float16* Vp = qkv + (size_t)kb * LDQ + 2 * HIDDEN + hq;
#pragma unroll
    for (int rep = 0; rep < 4; ++rep) {
      int idx = rep * 128 + threadIdx.x;  // 512 chunks of 8 halfs
      int row = idx >> 3;                 // key row 0..63
      int c0  = (idx & 7) * 8;            // d start
      v8h_t vv = *(const v8h_t*)(Vp + (size_t)row * LDQ + c0);
#pragma unroll
      for (int j = 0; j < 8; ++j) Vt[c0 + j][row] = vv[j];
    }
    __syncthreads();

    // S^T = K * Q^T : M = key(64, 4 tiles), N = query(16)
    const _Float16* Kp = qkv + (size_t)kb * LDQ + HIDDEN + hq;
    v8f_t st[4];
#pragma unroll
    for (int kt = 0; kt < 4; ++kt) {
      v8f_t s = {};
      v16h_t ka0 = load_a_frag(Kp + (size_t)(kt * 16) * LDQ + 0,  LDQ);
      v16h_t ka1 = load_a_frag(Kp + (size_t)(kt * 16) * LDQ + 32, LDQ);
      s = wmma_f16(ka0, bq0, s);
      s = wmma_f16(ka1, bq1, s);
      st[kt] = s;
    }

    // online softmax along keys (VGPRs + one lane-pair shuffle)
    float cmax = -1.0e30f;
#pragma unroll
    for (int kt = 0; kt < 4; ++kt)
#pragma unroll
      for (int r = 0; r < 8; ++r) cmax = fmaxf(cmax, st[kt][r] * scale);
    cmax = fmaxf(cmax, __shfl_xor(cmax, 16, 32));
    float m_new = fmaxf(m_i, cmax);
    float alpha = __expf(m_i - m_new);
    float rs = 0.0f;
#pragma unroll
    for (int kt = 0; kt < 4; ++kt)
#pragma unroll
      for (int r = 0; r < 8; ++r) {
        float p = __expf(st[kt][r] * scale - m_new);
        rs += p;
        Pt[w][qc][kt * 16 + rb + r] = (_Float16)p;  // transpose via LDS
      }
    rs += __shfl_xor(rs, 16, 32);
    l_i = l_i * alpha + rs;
    m_i = m_new;

    // rescale O (alpha broadcast: query q's stats live in lane q and q+16)
#pragma unroll
    for (int r = 0; r < 8; ++r) {
      float av = __shfl(alpha, rb + r, 32);
      o0[r] *= av; o1[r] *= av; o2[r] *= av; o3[r] *= av;
    }

    // O += P * V : A = P[16q x 32k] from LDS, B = V^T rows from LDS
#pragma unroll
    for (int ks = 0; ks < 2; ++ks) {
      v16h_t pa = load_a_frag(&Pt[w][0][0] + ks * 32, HDIM);
      o0 = wmma_f16(pa, load_b_frag(&Vt[ 0][0] + ks * 32, HDIM + 8), o0);
      o1 = wmma_f16(pa, load_b_frag(&Vt[16][0] + ks * 32, HDIM + 8), o1);
      o2 = wmma_f16(pa, load_b_frag(&Vt[32][0] + ks * 32, HDIM + 8), o2);
      o3 = wmma_f16(pa, load_b_frag(&Vt[48][0] + ks * 32, HDIM + 8), o3);
    }
    __syncthreads();
  }

  // finalize: /l and the reference's /(1+1e-8); store ctx[s][h*64+d] as f16
  const float post = 1.0f / (1.0f + 1e-8f);
#pragma unroll
  for (int r = 0; r < 8; ++r) {
    float lv  = __shfl(l_i, rb + r, 32);
    float inv = post / lv;
    int qrow = q0 + rb + r;
    _Float16* dst = ctx + (size_t)qrow * HIDDEN + hq;
    dst[ 0 + qc] = (_Float16)(o0[r] * inv);
    dst[16 + qc] = (_Float16)(o1[r] * inv);
    dst[32 + qc] = (_Float16)(o2[r] * inv);
    dst[48 + qc] = (_Float16)(o3[r] * inv);
  }
}

extern "C" void kernel_launch(void* const* d_in, const int* in_sizes, int n_in,
                              void* d_out, int out_size, void* d_ws, size_t ws_size,
                              hipStream_t stream) {
  const float* x    = (const float*)d_in[0];   // [4096,1024]
  const float* Wqkv = (const float*)d_in[1];   // [3072,1024]
  const float* Wout = (const float*)d_in[2];   // [1024,1024]
  float* out = (float*)d_out;                  // [4096,1024] f32

  char* ws = (char*)d_ws;
  _Float16* xh    = (_Float16*)(ws);
  _Float16* wqkvh = (_Float16*)(ws + (size_t) 8 * 1024 * 1024);
  _Float16* wouth = (_Float16*)(ws + (size_t)14 * 1024 * 1024);
  _Float16* qkvh  = (_Float16*)(ws + (size_t)16 * 1024 * 1024);
  _Float16* ctxh  = (_Float16*)(ws + (size_t)40 * 1024 * 1024);

  f32_to_f16_kernel<<<512, 256, 0, stream>>>(x,    xh,    SEQ * HIDDEN);
  f32_to_f16_kernel<<<512, 256, 0, stream>>>(Wqkv, wqkvh, 3 * HIDDEN * HIDDEN);
  f32_to_f16_kernel<<<512, 256, 0, stream>>>(Wout, wouth, HIDDEN * HIDDEN);

  // qkv = x @ Wqkv^T  -> f16 [4096,3072]
  gemm_xwt_wmma<<<dim3(3 * HIDDEN / 64, SEQ / 64), 128, 0, stream>>>(
      xh, wqkvh, qkvh, nullptr, HIDDEN, 3 * HIDDEN);

  // flash attention -> ctx f16 [4096,1024]
  flash_attn_wmma<<<dim3(SEQ / 64, HEADS), 128, 0, stream>>>(qkvh, ctxh);

  // y = ctx @ Wout^T -> f32 d_out
  gemm_xwt_wmma<<<dim3(HIDDEN / 64, SEQ / 64), 128, 0, stream>>>(
      ctxh, wouth, nullptr, out, HIDDEN, HIDDEN);
}
